// BinaryTreeLatentVariable_60516089200723
// MI455X (gfx1250) — matrix operand order repaired and checked by reading, not compiled, not run on hardware
//
#include <hip/hip_runtime.h>
#include <cmath>

// CDNA5 / gfx1250, wave32.
typedef __attribute__((ext_vector_type(16))) _Float16     v16h;
typedef __attribute__((ext_vector_type(8)))  _Float16     v8h;
typedef __attribute__((ext_vector_type(8)))  float        v8f;
typedef __attribute__((ext_vector_type(4)))  unsigned int v4u;
typedef __attribute__((ext_vector_type(8)))  int          v8i;
typedef __attribute__((ext_vector_type(4)))  int          v4i;

#define N_LEAVES 1024
#define S        64      // L*C = 16*4 states per node
#define K_TOT    4096    // S*S contracted (j,k) pairs

// ---------------------------------------------------------------------------
// Precompute Wt[i][jk] = exp(trans[l_p,l_l,l_r,c_p,c_l,c_r]) in f16, where
//   i = l_p*4+c_p, j = l_l*4+c_l, k = l_r*4+c_r, jk = j*64+k.
// Layout [64][4096] so a WMMA B-fragment lane reads two 16-byte runs of one row and a
// TDM tile (16 rows x 512B) covers one wave's N-tile for 8 K-chunks.
// ---------------------------------------------------------------------------
__global__ __launch_bounds__(256) void btlv_build_w_kernel(
    const float* __restrict__ trans, _Float16* __restrict__ Wt)
{
    int idx = blockIdx.x * 256 + threadIdx.x;      // 0 .. 262143
    int i  = idx >> 12;
    int jk = idx & 4095;
    int j  = jk >> 6;
    int k  = jk & 63;
    int lp = i >> 2, cp = i & 3;
    int ll = j >> 2, cl = j & 3;
    int lr = k >> 2, cr = k & 3;
    int src = ((lp * 16 + ll) * 16 + lr) * 64 + (cp * 4 + cl) * 4 + cr;
    Wt[idx] = (_Float16)expf(trans[src]);
}

// ---------------------------------------------------------------------------
// TDM: DMA 16 rows x 512 bytes (256 f16) of Wt into LDS.
// D# per CDNA5 ISA ch.8: group0 = {count, lds_addr, global_addr, type=2},
// group1 = {data_size=2B, tensor 256x16, tile 256x16, dim0_stride=4096}.
// Groups 2/3 zero (2D tensor). Tracked by this wave's TENSORcnt.
// ---------------------------------------------------------------------------
__device__ __forceinline__ void tdm_load_tile(const _Float16* gsrc,
                                              const _Float16* ldst)
{
    unsigned long long ga = (unsigned long long)(uintptr_t)gsrc;
    unsigned int       la = (unsigned int)(uintptr_t)ldst;  // low 32b = LDS offset

    v4u g0;
    g0[0] = 1u;                                             // count=1 valid D#
    g0[1] = la;                                             // lds_addr  [63:32]
    g0[2] = (unsigned int)ga;                               // global_addr[95:64]
    g0[3] = (unsigned int)((ga >> 32) & 0x01FFFFFFu)        // global_addr[120:96]
          | (2u << 30);                                     // type=2 "image"

    v8i g1;
    g1[0] = (int)(1u << 16);        // data_size=1 (2 bytes); wg_mask=0
    g1[1] = (int)(256u << 16);      // tensor_dim0 = 256 elems   (bits 63:48)
    g1[2] = (int)(16u << 16);       // tensor_dim1 = 16 rows     (bits 95:80)
    g1[3] = (int)(256u << 16);      // tile_dim0   = 256 elems   (bits 127:112)
    g1[4] = 16;                     // tile_dim1   = 16 rows     (bits 143:128)
    g1[5] = 4096;                   // tensor_dim0_stride = 4096 elems
    g1[6] = 0;
    g1[7] = 0;

    v4i z4 = {0, 0, 0, 0};
    v8i z8 = {0, 0, 0, 0, 0, 0, 0, 0};
    __builtin_amdgcn_tensor_load_to_lds(g0, g1, z4, z4, z8, 0);
}

// ---------------------------------------------------------------------------
// One tree level: n parent nodes.
//   out[b][i] = parent[b][i] + maxL[b] + maxR[b]
//             + log( sum_{j,k} eL[b][j] * Wt[i][j*64+k] * eR[b][k] )
// One block = 16 nodes; wave w (of 4) owns output columns i = w*16..w*16+15,
// so its B-operand is exactly Wt rows [w*16, w*16+16) — streamed by TDM into
// a wave-private double-buffered LDS stage, consumed via ds_load_b128 into
// v_wmma_f32_16x16x32_f16. Two independent accumulator chains (even/odd
// K-chunk) break the XDL RAW dependence at single-wave occupancy.
// ---------------------------------------------------------------------------
__global__ __launch_bounds__(128) void btlv_level_kernel(
    const float*    __restrict__ child,
    const float*    __restrict__ parent,
    const _Float16* __restrict__ Wt,
    float*          __restrict__ outp,
    int n)
{
    __shared__ __align__(16) _Float16 eStage[2][16][64];       // [L/R][node][state]
    __shared__ float mStage[2][16];
    __shared__ __align__(16) _Float16 bStage[4][2][16][256];   // [wave][buf][row][k] 64KB

    const int tid   = threadIdx.x;
    const int lane  = tid & 31;
    const int wave  = __builtin_amdgcn_readfirstlane(tid >> 5);
    const int tbase = blockIdx.x * 16;
    const int valid = (n - tbase < 16) ? (n - tbase) : 16;

    // ---- Stage eL/eR: threads 0..31 each handle one child row ----
    if (tid < 32) {
        const int b    = tid & 15;      // local node
        const int side = tid >> 4;      // 0 = left, 1 = right
        if (b < valid) {
            const float* row = child + (size_t)(2 * (tbase + b) + side) * S;
            float mx = row[0];
            #pragma unroll
            for (int q = 1; q < S; ++q) mx = fmaxf(mx, row[q]);
            #pragma unroll
            for (int q = 0; q < S; ++q)
                eStage[side][b][q] = (_Float16)expf(row[q] - mx);
            mStage[side][b] = mx;
        } else {
            #pragma unroll
            for (int q = 0; q < S; ++q) eStage[side][b][q] = (_Float16)0.0f;
            mStage[side][b] = 0.0f;
        }
    }
    __syncthreads();

    // ---- WMMA pipeline, wave-private TDM double buffer ----
    const int m    = lane & 15;     // A row (node) / B-D column
    const int hi   = lane >> 4;
    const int khA  = hi * 8;        // A-frag K sub-offset (16-bit A 16x32 layout)
    const int khB  = hi * 8;        // B-frag K sub-offset (same table order as A)

    const _Float16* eLrow = &eStage[0][m][0];
    const _Float16* eRrow = &eStage[1][m][0];
    const _Float16* wrows = Wt + (size_t)wave * 16 * K_TOT;   // this wave's 16 rows

    v8f acc0 = (v8f){};
    v8f acc1 = (v8f){};

    tdm_load_tile(wrows, &bStage[wave][0][0][0]);             // prologue: stage 0
    for (int s = 0; s < 16; ++s) {
        if (s < 15) {
            tdm_load_tile(wrows + (size_t)(s + 1) * 256,
                          &bStage[wave][(s + 1) & 1][0][0]);
            __builtin_amdgcn_s_wait_tensorcnt(1);             // stage s landed
        } else {
            __builtin_amdgcn_s_wait_tensorcnt(0);
        }
        asm volatile("" ::: "memory");                        // LDS written by TDM

        const _Float16* bb = &bStage[wave][s & 1][m][0];      // row m = lane&15
        #pragma unroll
        for (int kl = 0; kl < 8; ++kl) {
            const int kk = s * 8 + kl;
            const int j0 = kk >> 1;               // jk = j*64+k, fixed j per chunk
            const int kb = (kk & 1) * 32;
            const _Float16 sc = eLrow[j0];

            v8h r0 = *(const v8h*)(eRrow + kb + khA);         // ds_load_b128
            v8h r1 = *(const v8h*)(eRrow + kb + 16 + khA);
            v16h a;
            #pragma unroll
            for (int p = 0; p < 8; ++p) { a[p] = sc * r0[p]; a[8 + p] = sc * r1[p]; }

            v8h b0 = *(const v8h*)(bb + kl * 32 + khB);       // ds_load_b128
            v8h b1 = *(const v8h*)(bb + kl * 32 + khB + 16);
            v16h bf;
            #pragma unroll
            for (int p = 0; p < 8; ++p) { bf[p] = b0[p]; bf[8 + p] = b1[p]; }

            if (kl & 1)
                acc1 = __builtin_amdgcn_wmma_f32_16x16x32_f16(
                    false, a, false, bf, (short)0, acc1, false, false);
            else
                acc0 = __builtin_amdgcn_wmma_f32_16x16x32_f16(
                    false, a, false, bf, (short)0, acc0, false, false);
        }
    }

    // ---- Epilogue: D VGPR r -> row r + hi*8, col = lane&15; i = wave*16+col ----
    #pragma unroll
    for (int r = 0; r < 8; ++r) {
        const int mrow = r + hi * 8;
        if (mrow < valid) {
            const int node = tbase + mrow;
            const int i    = wave * 16 + m;
            const float ms = mStage[0][mrow] + mStage[1][mrow];
            outp[(size_t)node * S + i] =
                parent[(size_t)node * S + i] + ms + logf(acc0[r] + acc1[r]);
        }
    }
}

// ---------------------------------------------------------------------------
// Launch: 1 precompute + 10 chained level kernels (stream-ordered).
// Workspace: bufA (128KB) | bufB (128KB) | Wt (512KB f16)  =>  768KB of d_ws.
// ---------------------------------------------------------------------------
extern "C" void kernel_launch(void* const* d_in, const int* in_sizes, int n_in,
                              void* d_out, int out_size, void* d_ws, size_t ws_size,
                              hipStream_t stream)
{
    const float* node_state = (const float*)d_in[0];   // (2047, 16, 4) f32
    const float* trans      = (const float*)d_in[1];   // (16,16,16,4,4,4) f32
    float*       out        = (float*)d_out;           // (16, 4) f32

    char* ws = (char*)d_ws;
    float*    bufA = (float*)(ws);
    float*    bufB = (float*)(ws + (size_t)512 * S * sizeof(float));
    _Float16* Wt   = (_Float16*)(ws + (size_t)2 * 512 * S * sizeof(float));

    btlv_build_w_kernel<<<K_TOT * S / 256, 256, 0, stream>>>(trans, Wt);

    const float* child = node_state + (size_t)(N_LEAVES - 1) * S;  // leaves
    float* pong[2] = {bufA, bufB};
    int pb = 0;
    for (int n = N_LEAVES / 2; n >= 1; n >>= 1) {
        float* o = (n == 1) ? out : pong[pb];
        int blocks = (n + 15) / 16;
        btlv_level_kernel<<<blocks, 128, 0, stream>>>(
            child, node_state + (size_t)(n - 1) * S, Wt, o, n);
        child = o;
        pb ^= 1;
    }
}
